// GroupedLinear_25443386262127
// MI455X (gfx1250) — matrix operand order repaired and checked by reading, not compiled, not run on hardware
//
#include <hip/hip_runtime.h>

// CDNA5 / gfx1250 vector types for WMMA
typedef __attribute__((ext_vector_type(16))) __bf16 v16bf;
typedef __attribute__((ext_vector_type(8)))  float  v8f;

static constexpr int G_GRP = 16;
static constexpr int NTOK  = 32768;
static constexpr int KDIM  = 1024;   // IN
static constexpr int ODIM  = 1024;   // OUT

static constexpr int BM = 128;       // block rows (tokens)
static constexpr int BN = 128;       // block cols (out features)
static constexpr int BK = 32;        // K chunk (one bf16 WMMA K)
static constexpr int LDSS = 40;      // LDS row stride in ushorts (32 + 8 pad -> conflict-free)

__device__ __forceinline__ unsigned short bf16_rne(float f) {
  unsigned u = __float_as_uint(f);
  u += 0x7FFFu + ((u >> 16) & 1u);
  return (unsigned short)(u >> 16);
}

__device__ __forceinline__ void bf16_split(float f, unsigned short& h, unsigned short& l) {
  h = bf16_rne(f);
  float hf = __uint_as_float(((unsigned)h) << 16);
  l = bf16_rne(f - hf);
}

union Frag { uint4 q[2]; v16bf v; };

__global__ __launch_bounds__(256)
void grouped_gemm_bf16x3(const float* __restrict__ x,
                         const float* __restrict__ w,
                         const int*   __restrict__ glens,
                         float*       __restrict__ out) {
  __shared__ __align__(16) unsigned short sAh[BM * LDSS];
  __shared__ __align__(16) unsigned short sAl[BM * LDSS];
  __shared__ __align__(16) unsigned short sBh[BN * LDSS];
  __shared__ __align__(16) unsigned short sBl[BN * LDSS];

  const int tid  = threadIdx.x;
  const int lane = tid & 31;
  const int wave = tid >> 5;          // 8 waves
  const int wm   = wave >> 1;         // 0..3 -> 4 * 32 rows = 128
  const int wn   = wave & 1;          // 0..1 -> 2 * 64 cols = 128
  const int row0 = (int)blockIdx.y * BM;
  const int col0 = (int)blockIdx.x * BN;

  // 2 (M) x 4 (N) accumulator tiles of 16x16 f32
  v8f acc[2][4];
  #pragma unroll
  for (int i = 0; i < 2; ++i)
    #pragma unroll
    for (int j = 0; j < 4; ++j)
      acc[i][j] = v8f{};

  int edge = 0;
  for (int g = 0; g < G_GRP; ++g) {
    const int gLo = edge;
    edge += glens[g];                 // uniform across block
    const int gHi = edge;
    const int lo = gLo > row0 ? gLo : row0;
    const int hi = gHi < row0 + BM ? gHi : row0 + BM;
    if (lo >= hi) continue;           // group doesn't touch this row block

    const float* __restrict__ wg = w + (size_t)g * ODIM * KDIM;

    for (int k0 = 0; k0 < KDIM; k0 += BK) {
      // ---- stage A (masked to [lo,hi)) and B into LDS, split hi/lo bf16 ----
      #pragma unroll
      for (int it = 0; it < 4; ++it) {
        const int idx = tid + it * 256;     // 0..1023
        const int r   = idx >> 3;           // 0..127
        const int c4  = (idx & 7) << 2;     // 0,4,...,28
        // A: x rows, zero outside the group segment
        const int grow = row0 + r;
        float4 av = make_float4(0.f, 0.f, 0.f, 0.f);
        if (grow >= lo && grow < hi)
          av = *(const float4*)(x + (size_t)grow * KDIM + (k0 + c4));
        unsigned short h0,h1,h2,h3,l0,l1,l2,l3;
        bf16_split(av.x, h0, l0); bf16_split(av.y, h1, l1);
        bf16_split(av.z, h2, l2); bf16_split(av.w, h3, l3);
        *(uint2*)&sAh[r * LDSS + c4] =
            make_uint2((unsigned)h0 | ((unsigned)h1 << 16), (unsigned)h2 | ((unsigned)h3 << 16));
        *(uint2*)&sAl[r * LDSS + c4] =
            make_uint2((unsigned)l0 | ((unsigned)l1 << 16), (unsigned)l2 | ((unsigned)l3 << 16));
        // B: weight[g][col0+r][k0+c4..+3] (B^T of the GEMM; contiguous along K)
        float4 bv = *(const float4*)(wg + (size_t)(col0 + r) * KDIM + (k0 + c4));
        bf16_split(bv.x, h0, l0); bf16_split(bv.y, h1, l1);
        bf16_split(bv.z, h2, l2); bf16_split(bv.w, h3, l3);
        *(uint2*)&sBh[r * LDSS + c4] =
            make_uint2((unsigned)h0 | ((unsigned)h1 << 16), (unsigned)h2 | ((unsigned)h3 << 16));
        *(uint2*)&sBl[r * LDSS + c4] =
            make_uint2((unsigned)l0 | ((unsigned)l1 << 16), (unsigned)l2 | ((unsigned)l3 << 16));
      }
      __syncthreads();

      // ---- fragments + WMMA ----
      // A layout (16-bit, 16x32): lanes 0-15 = row m, K{0..7,16..23}; lanes 16-31 = K{8..15,24..31}
      Frag aH[2], aL[2];
      #pragma unroll
      for (int mi = 0; mi < 2; ++mi) {
        const int arow = wm * 32 + mi * 16 + (lane & 15);
        const int kb   = (lane >> 4) * 8;
        const unsigned short* pa = &sAh[arow * LDSS + kb];
        aH[mi].q[0] = *(const uint4*)(pa);
        aH[mi].q[1] = *(const uint4*)(pa + 16);
        const unsigned short* pl = &sAl[arow * LDSS + kb];
        aL[mi].q[0] = *(const uint4*)(pl);
        aL[mi].q[1] = *(const uint4*)(pl + 16);
      }
      #pragma unroll
      for (int nj = 0; nj < 4; ++nj) {
        // B layout (16-bit, 32x16): lanes 0-15 = col n, K{0..7,16..23}; lanes 16-31 = K{8..15,24..31}
        const int brow = wn * 64 + nj * 16 + (lane & 15);
        const int kb   = (lane >> 4) * 8;
        Frag bH, bL;
        const unsigned short* pb = &sBh[brow * LDSS + kb];
        bH.q[0] = *(const uint4*)(pb);
        bH.q[1] = *(const uint4*)(pb + 16);
        const unsigned short* pc = &sBl[brow * LDSS + kb];
        bL.q[0] = *(const uint4*)(pc);
        bL.q[1] = *(const uint4*)(pc + 16);
        #pragma unroll
        for (int mi = 0; mi < 2; ++mi) {
          // fp32-accurate product via 3 bf16 WMMAs: hi*hi + hi*lo + lo*hi
          acc[mi][nj] = __builtin_amdgcn_wmma_f32_16x16x32_bf16(
              false, aH[mi].v, false, bH.v, (short)0, acc[mi][nj], false, false);
          acc[mi][nj] = __builtin_amdgcn_wmma_f32_16x16x32_bf16(
              false, aH[mi].v, false, bL.v, (short)0, acc[mi][nj], false, false);
          acc[mi][nj] = __builtin_amdgcn_wmma_f32_16x16x32_bf16(
              false, aL[mi].v, false, bH.v, (short)0, acc[mi][nj], false, false);
        }
      }
      __syncthreads();
    }
  }

  // ---- store C: 16x16 f32 tile layout: lanes 0-15 -> M=v, lanes 16-31 -> M=8+v ----
  #pragma unroll
  for (int mi = 0; mi < 2; ++mi) {
    #pragma unroll
    for (int nj = 0; nj < 4; ++nj) {
      const int rbase = row0 + wm * 32 + mi * 16 + ((lane >> 4) << 3);
      const int c     = col0 + wn * 64 + nj * 16 + (lane & 15);
      #pragma unroll
      for (int v = 0; v < 8; ++v)
        out[(size_t)(rbase + v) * ODIM + c] = acc[mi][nj][v];
    }
  }
}

extern "C" void kernel_launch(void* const* d_in, const int* in_sizes, int n_in,
                              void* d_out, int out_size, void* d_ws, size_t ws_size,
                              hipStream_t stream) {
  const float* x  = (const float*)d_in[0];   // [N, IN] fp32
  const float* w  = (const float*)d_in[1];   // [G, OUT, IN] fp32
  const int*   gl = (const int*)d_in[2];     // [G] int32, sums to N
  float*       o  = (float*)d_out;           // [N, OUT] fp32
  (void)in_sizes; (void)n_in; (void)out_size; (void)d_ws; (void)ws_size;

  dim3 grid(ODIM / BN, NTOK / BM);           // (8, 256)
  grouped_gemm_bf16x3<<<grid, 256, 0, stream>>>(x, w, gl, o);
}
